// SparseWindowVectorizedTransformerEncoderLayer_40724879900950
// MI455X (gfx1250) — matrix operand order repaired and checked by reading, not compiled, not run on hardware
//
#include <hip/hip_runtime.h>

#define S_LEN 4096
#define DMODEL 512
#define NHEAD 8
#define HDIM 64
#define WIN 128
#define DFF 2048
#define QKV_LD (3 * DMODEL)

typedef __attribute__((ext_vector_type(16))) __bf16 v16bf;
typedef __attribute__((ext_vector_type(8)))  __bf16 v8bf;
typedef __attribute__((ext_vector_type(8)))  float  v8f;
typedef __attribute__((ext_vector_type(4)))  float  v4f;

union BF16x16 { v16bf v; v8bf h[2]; };

__device__ __forceinline__ __bf16 f2bf(float f) {
  union { float f; unsigned u; } in; in.f = f;
  unsigned u = in.u;
  u += 0x7FFFu + ((u >> 16) & 1u);           // round-to-nearest-even
  union { unsigned short s; __bf16 b; } out;
  out.s = (unsigned short)(u >> 16);
  return out.b;
}

// A-fragment (16x32 bf16, MxK): lane holds row (lane&15); K chunks at
// h*8..h*8+7 and 16+h*8..16+h*8+7 where h = lane>>4. Caller pre-offsets
// pointer by row*ld + (lane>>4)*8; we read +0 and +16.
__device__ __forceinline__ v16bf load_frag_a(const __bf16* p) {
  BF16x16 f;
  f.h[0] = *(const v8bf*)(p);
  f.h[1] = *(const v8bf*)(p + 16);
  return f.v;
}

// B-fragment (32x16 bf16, KxN) loaded from B^T [N,K]: lane holds column
// (lane&15); K chunks at h*8..h*8+7 and 16+h*8..16+h*8+7. Caller pre-offsets
// pointer by col*ld + (lane>>4)*8; we read +0 and +16.
__device__ __forceinline__ v16bf load_frag_b(const __bf16* p) {
  BF16x16 f;
  f.h[0] = *(const v8bf*)(p);
  f.h[1] = *(const v8bf*)(p + 16);
  return f.v;
}

__device__ __forceinline__ v8f wmma_bf16(v16bf a, v16bf b, v8f c) {
  return __builtin_amdgcn_wmma_f32_16x16x32_bf16(false, a, false, b,
                                                 (short)0, c, false, false);
}

// ---------------- conversion kernels ----------------

__global__ void cvt_bf16_kernel(const float* __restrict__ in,
                                __bf16* __restrict__ out, int n) {
  int i = blockIdx.x * blockDim.x + threadIdx.x;
  if (i < n) out[i] = f2bf(in[i]);
}

// w is [K,N] fp32 (in,out layout); wt is [N,K] bf16 (transposed)
__global__ void cvt_t_kernel(const float* __restrict__ w,
                             __bf16* __restrict__ wt, int K, int N) {
  int i = blockIdx.x * blockDim.x + threadIdx.x;
  if (i < N * K) {
    int n = i / K;
    int k = i - n * K;
    wt[i] = f2bf(w[(size_t)k * N + n]);
  }
}

// ---------------- GEMM: C = A[MxK] * Bt[NxK]^T + bias ----------------
// mode 0: store bf16; mode 1: store bf16 with ReLU; mode 2: store f32 + resid
__global__ __launch_bounds__(256) void gemm_bf16_kernel(
    const __bf16* __restrict__ A, const __bf16* __restrict__ Bt,
    const float* __restrict__ bias, const float* __restrict__ resid,
    float* __restrict__ outf, __bf16* __restrict__ outb,
    int M, int N, int K, int mode) {
  (void)M;
  const int lane = threadIdx.x & 31;
  const int wave = threadIdx.x >> 5;
  const int wm = wave >> 1;           // 0..3
  const int wn = wave & 1;            // 0..1
  const int hh = lane >> 4;           // half-wave selector
  const int ln = lane & 15;
  const int m0 = blockIdx.y * 128 + wm * 32;
  const int n0 = blockIdx.x * 128 + wn * 64;

  const __bf16* Ab = A + (size_t)(m0 + ln) * K + hh * 8;
  const __bf16* Bb = Bt + (size_t)(n0 + ln) * K + hh * 8;

  v8f acc[2][4] = {};

  for (int k = 0; k < K; k += 32) {
    v16bf a0 = load_frag_a(Ab + k);
    v16bf a1 = load_frag_a(Ab + (size_t)16 * K + k);
    v16bf b0 = load_frag_b(Bb + k);
    v16bf b1 = load_frag_b(Bb + (size_t)16 * K + k);
    v16bf b2 = load_frag_b(Bb + (size_t)32 * K + k);
    v16bf b3 = load_frag_b(Bb + (size_t)48 * K + k);
    acc[0][0] = wmma_bf16(a0, b0, acc[0][0]);
    acc[0][1] = wmma_bf16(a0, b1, acc[0][1]);
    acc[0][2] = wmma_bf16(a0, b2, acc[0][2]);
    acc[0][3] = wmma_bf16(a0, b3, acc[0][3]);
    acc[1][0] = wmma_bf16(a1, b0, acc[1][0]);
    acc[1][1] = wmma_bf16(a1, b1, acc[1][1]);
    acc[1][2] = wmma_bf16(a1, b2, acc[1][2]);
    acc[1][3] = wmma_bf16(a1, b3, acc[1][3]);
  }

  for (int mi = 0; mi < 2; ++mi) {
    for (int ni = 0; ni < 4; ++ni) {
      int n = n0 + ni * 16 + ln;
      float bv = bias[n];
      for (int r = 0; r < 8; ++r) {
        int m = m0 + mi * 16 + hh * 8 + r;
        float v = acc[mi][ni][r] + bv;
        size_t off = (size_t)m * N + n;
        if (mode == 2) {
          outf[off] = v + resid[off];
        } else {
          if (mode == 1) v = fmaxf(v, 0.0f);
          outb[off] = f2bf(v);
        }
      }
    }
  }
}

// ---------------- windowed attention ----------------
// qkvb: [S, 3*D] bf16 (q|k|v per row). ob: [S, D] bf16 attention output.
#define KPAD 288   // max window keys (272) padded to multiple of 32
#define PSTR 292   // f32 score/prob row stride (bank-conflict padding)
#define VSTR 296   // bf16 V^T row stride

__global__ __launch_bounds__(128) void attn_kernel(
    const __bf16* __restrict__ qkvb, __bf16* __restrict__ ob) {
  __shared__ float sc[16 * PSTR];          // scores, then probabilities
  __shared__ __bf16 vt[HDIM * VSTR];       // V tile transposed [col][key]
  __shared__ float red1[128];
  __shared__ float red2[128];

  const int q0 = blockIdx.x * 16;
  const int h = blockIdx.y;
  const int tid = threadIdx.x;
  const int lane = tid & 31;
  const int wave = tid >> 5;
  const int hh = lane >> 4;
  const int ln = lane & 15;

  const int kstart = (q0 - WIN) > 0 ? (q0 - WIN) : 0;
  const int kend = (q0 + 16 + WIN) < S_LEN ? (q0 + 16 + WIN) : S_LEN;
  const int klen = kend - kstart;
  const int ntiles = (klen + 15) >> 4;

  // Q fragment (shared by all score tiles of this wave)
  const __bf16* qbase = qkvb + (size_t)(q0 + ln) * QKV_LD + h * HDIM + hh * 8;
  v16bf aq0 = load_frag_a(qbase + 0);
  v16bf aq1 = load_frag_a(qbase + 32);

  // ---- phase 1: scores = Q K^T / sqrt(d), masked to the band ----
  const float scale = 0.04419417382415922f;  // 1/sqrt(512)
  for (int kt = wave; kt < ntiles; kt += 4) {
    int krow = kstart + kt * 16 + ln;
    int krc = krow < S_LEN ? krow : (S_LEN - 1);
    const __bf16* kbase =
        qkvb + (size_t)krc * QKV_LD + DMODEL + h * HDIM + hh * 8;
    v8f acc = {};
    acc = wmma_bf16(aq0, load_frag_b(kbase), acc);
    acc = wmma_bf16(aq1, load_frag_b(kbase + 32), acc);
    for (int r = 0; r < 8; ++r) {
      int m = hh * 8 + r;
      int key = kstart + kt * 16 + ln;
      int d = (q0 + m) - key;
      bool valid = (key < S_LEN) && (d <= WIN) && (d >= -WIN);
      sc[m * PSTR + kt * 16 + ln] = valid ? acc[r] * scale : -1e30f;
    }
  }

  // ---- stage V^T into LDS (zero-pad past klen: 0 * anything stays 0) ----
  const int voff = 2 * DMODEL + h * HDIM;
  const __bf16 bzero = f2bf(0.0f);
  for (int idx = tid; idx < HDIM * KPAD; idx += 128) {
    int c = idx & (HDIM - 1);
    int kk = idx >> 6;
    __bf16 val = bzero;
    if (kk < klen)
      val = qkvb[(size_t)(kstart + kk) * QKV_LD + voff + c];
    vt[c * VSTR + kk] = val;
  }
  __syncthreads();

  // ---- phase 2: softmax over each row's window, in place (f32) ----
  const int row = tid >> 3;   // 16 rows x 8 threads
  const int sub = tid & 7;
  float mx = -1e30f;
  for (int k = sub; k < klen; k += 8) mx = fmaxf(mx, sc[row * PSTR + k]);
  red1[tid] = mx;
  __syncthreads();
  float rowmax = -1e30f;
  for (int i = 0; i < 8; ++i) rowmax = fmaxf(rowmax, red1[row * 8 + i]);
  float sm = 0.0f;
  for (int k = sub; k < klen; k += 8)
    sm += __expf(sc[row * PSTR + k] - rowmax);
  red2[tid] = sm;
  __syncthreads();
  float rowsum = 0.0f;
  for (int i = 0; i < 8; ++i) rowsum += red2[row * 8 + i];
  float inv = 1.0f / rowsum;
  for (int k = sub; k < KPAD; k += 8) {
    float p = 0.0f;
    if (k < klen) p = __expf(sc[row * PSTR + k] - rowmax) * inv;
    sc[row * PSTR + k] = p;
  }
  __syncthreads();

  // ---- phase 3: O = P V via WMMA; each wave handles 16 output columns ----
  const int n0 = wave * 16;
  v8f acc = {};
  for (int ks = 0; ks < KPAD; ks += 32) {
    const float* prow = &sc[ln * PSTR + ks + hh * 8];
    v4f p0 = *(const v4f*)(prow);
    v4f p1 = *(const v4f*)(prow + 4);
    v4f p2 = *(const v4f*)(prow + 16);
    v4f p3 = *(const v4f*)(prow + 20);
    BF16x16 af;
    for (int j = 0; j < 4; ++j) {
      af.v[j] = f2bf(p0[j]);
      af.v[4 + j] = f2bf(p1[j]);
      af.v[8 + j] = f2bf(p2[j]);
      af.v[12 + j] = f2bf(p3[j]);
    }
    const __bf16* vrow = &vt[(n0 + ln) * VSTR + ks + hh * 8];
    BF16x16 fb;
    fb.h[0] = *(const v8bf*)(vrow);
    fb.h[1] = *(const v8bf*)(vrow + 16);
    acc = wmma_bf16(af.v, fb.v, acc);
  }
  for (int r = 0; r < 8; ++r) {
    int m = hh * 8 + r;
    ob[(size_t)(q0 + m) * DMODEL + h * HDIM + n0 + ln] = f2bf(acc[r]);
  }
}

// ---------------- layernorm (row = 512 floats) ----------------
__global__ __launch_bounds__(128) void ln_kernel(
    const float* __restrict__ in, const float* __restrict__ g,
    const float* __restrict__ b, float* __restrict__ outf,
    __bf16* __restrict__ outb) {
  __shared__ float s1[128];
  __shared__ float s2[128];
  const int row = blockIdx.x;
  const int tid = threadIdx.x;
  const float* x = in + (size_t)row * DMODEL;
  float v[4];
  float lsum = 0.0f, lsq = 0.0f;
  for (int i = 0; i < 4; ++i) {
    v[i] = x[tid + i * 128];
    lsum += v[i];
    lsq += v[i] * v[i];
  }
  s1[tid] = lsum;
  s2[tid] = lsq;
  __syncthreads();
  for (int s = 64; s > 0; s >>= 1) {
    if (tid < s) { s1[tid] += s1[tid + s]; s2[tid] += s2[tid + s]; }
    __syncthreads();
  }
  float mean = s1[0] * (1.0f / DMODEL);
  float var = s2[0] * (1.0f / DMODEL) - mean * mean;
  float rstd = rsqrtf(var + 1e-5f);
  for (int i = 0; i < 4; ++i) {
    int c = tid + i * 128;
    float y = (v[i] - mean) * rstd * g[c] + b[c];
    outf[(size_t)row * DMODEL + c] = y;
    if (outb) outb[(size_t)row * DMODEL + c] = f2bf(y);
  }
}

// ---------------- host orchestration ----------------

extern "C" void kernel_launch(void* const* d_in, const int* in_sizes, int n_in,
                              void* d_out, int out_size, void* d_ws,
                              size_t ws_size, hipStream_t stream) {
  (void)in_sizes; (void)n_in; (void)out_size; (void)ws_size;
  const float* x     = (const float*)d_in[0];
  const float* qkv_w = (const float*)d_in[1];
  const float* qkv_b = (const float*)d_in[2];
  const float* out_w = (const float*)d_in[3];
  const float* out_b = (const float*)d_in[4];
  const float* ln1_g = (const float*)d_in[5];
  const float* ln1_b = (const float*)d_in[6];
  const float* w1    = (const float*)d_in[7];
  const float* b1    = (const float*)d_in[8];
  const float* w2    = (const float*)d_in[9];
  const float* b2    = (const float*)d_in[10];
  const float* ln2_g = (const float*)d_in[11];
  const float* ln2_b = (const float*)d_in[12];

  char* ws = (char*)d_ws;
  size_t off = 0;
  auto alloc = [&](size_t bytes) -> void* {
    void* p = ws + off;
    off = (off + bytes + 255) & ~(size_t)255;
    return p;
  };

  const size_t S = S_LEN, D = DMODEL, F = DFF;
  __bf16* xb    = (__bf16*)alloc(S * D * 2);
  __bf16* qkvwT = (__bf16*)alloc(3 * D * D * 2);
  __bf16* outwT = (__bf16*)alloc(D * D * 2);
  __bf16* w1T   = (__bf16*)alloc(F * D * 2);
  __bf16* w2T   = (__bf16*)alloc(D * F * 2);
  __bf16* qkvb  = (__bf16*)alloc(S * 3 * D * 2);
  __bf16* obuf  = (__bf16*)alloc(S * D * 2);
  float*  ybuf  = (float*)alloc(S * D * 4);
  float*  x1    = (float*)alloc(S * D * 4);
  __bf16* x1b   = (__bf16*)alloc(S * D * 2);
  __bf16* hb    = (__bf16*)alloc(S * F * 2);

  int n;
  n = (int)(S * D);
  cvt_bf16_kernel<<<(n + 255) / 256, 256, 0, stream>>>(x, xb, n);
  n = (int)(3 * D * D);
  cvt_t_kernel<<<(n + 255) / 256, 256, 0, stream>>>(qkv_w, qkvwT, (int)D, (int)(3 * D));
  n = (int)(D * D);
  cvt_t_kernel<<<(n + 255) / 256, 256, 0, stream>>>(out_w, outwT, (int)D, (int)D);
  n = (int)(F * D);
  cvt_t_kernel<<<(n + 255) / 256, 256, 0, stream>>>(w1, w1T, (int)D, (int)F);
  n = (int)(D * F);
  cvt_t_kernel<<<(n + 255) / 256, 256, 0, stream>>>(w2, w2T, (int)F, (int)D);

  // QKV projection: [4096,512] x [512,1536] -> bf16 [4096,1536]
  gemm_bf16_kernel<<<dim3(3 * D / 128, S / 128), 256, 0, stream>>>(
      xb, qkvwT, qkv_b, nullptr, nullptr, qkvb, (int)S, (int)(3 * D), (int)D, 0);

  // windowed attention -> bf16 [4096,512]
  attn_kernel<<<dim3(S / 16, NHEAD), 128, 0, stream>>>(qkvb, obuf);

  // out-proj + residual(x) -> f32 ybuf
  gemm_bf16_kernel<<<dim3(D / 128, S / 128), 256, 0, stream>>>(
      obuf, outwT, out_b, x, ybuf, nullptr, (int)S, (int)D, (int)D, 2);

  // LN1 -> x1 (f32) + x1b (bf16)
  ln_kernel<<<S, 128, 0, stream>>>(ybuf, ln1_g, ln1_b, x1, x1b);

  // FF1 + ReLU -> bf16 [4096,2048]
  gemm_bf16_kernel<<<dim3(F / 128, S / 128), 256, 0, stream>>>(
      x1b, w1T, b1, nullptr, nullptr, hb, (int)S, (int)F, (int)D, 1);

  // FF2 + residual(x1) -> f32 ybuf
  gemm_bf16_kernel<<<dim3(D / 128, S / 128), 256, 0, stream>>>(
      hb, w2T, b2, x1, ybuf, nullptr, (int)S, (int)D, (int)F, 2);

  // LN2 -> final f32 output
  ln_kernel<<<S, 128, 0, stream>>>(ybuf, ln2_g, ln2_b, (float*)d_out, nullptr);
}
